// SAM2MultiScaleAttention_55405078119084
// MI455X (gfx1250) — matrix operand order repaired and checked by reading, not compiled, not run on hardware
//
#include <hip/hip_runtime.h>
#include <math.h>
#include <stdint.h>

typedef __bf16 bf16_t;
typedef __attribute__((ext_vector_type(16))) __bf16 v16bf;
typedef __attribute__((ext_vector_type(8)))  __bf16 v8bf;
typedef __attribute__((ext_vector_type(8)))  float   v8f;

#define L2E 1.44269504088896340736f

static __device__ inline v8f wmma_bf16(v16bf a, v16bf b, v8f c) {
  return __builtin_amdgcn_wmma_f32_16x16x32_bf16(false, a, false, b, (short)0, c, false, false);
}

// A-fragment (16x32 bf16, lane m = lane&15, half = lane>>4):
// dims {kb+half*16 .. +7} in r[0..7], dims {kb+half*16+8 .. +7} in r[8..15]
static __device__ inline v16bf a_frag_bf16(const bf16_t* lo, const bf16_t* hi) {
  v8bf l = *(const v8bf*)lo;
  v8bf h = *(const v8bf*)hi;
  v16bf r;
#pragma unroll
  for (int i = 0; i < 8; ++i) { r[i] = l[i]; r[8 + i] = h[i]; }
  return r;
}

// CDNA5 async global->LDS copy, 16B per lane, tracked by ASYNCcnt.
static __device__ inline void async_copy16(const bf16_t* lds_dst, const bf16_t* gsrc) {
  uint32_t laddr = (uint32_t)(uintptr_t)lds_dst;   // low 32 bits = LDS offset
  uint64_t gaddr = (uint64_t)(uintptr_t)gsrc;
  asm volatile("global_load_async_to_lds_b128 %0, %1, off"
               :: "v"(laddr), "v"(gaddr) : "memory");
}
static __device__ inline void wait_async_le3() {
  asm volatile("s_wait_asynccnt 0x3" ::: "memory");
}
static __device__ inline void wait_async_0() {
  asm volatile("s_wait_asynccnt 0x0" ::: "memory");
}

// ---------------- hs fp32 -> bf16 --------------------------------------
__global__ __launch_bounds__(256)
void k_cvt_bf16(const float* __restrict__ src, bf16_t* __restrict__ dst, int n) {
  int idx = blockIdx.x * 256 + threadIdx.x;
  if (idx < n) dst[idx] = (__bf16)src[idx];
}

// ---------------- weight transpose + bf16 convert: w[K][N] -> wT[N][K] ----
__global__ __launch_bounds__(256)
void k_transpose_bf16(const float* __restrict__ w, bf16_t* __restrict__ wT,
                      int K, int N) {
  int idx = blockIdx.x * 256 + threadIdx.x;
  if (idx >= K * N) return;
  int n = idx / K, k = idx - n * K;
  wT[idx] = (__bf16)w[(size_t)k * N + n];
}

// ---------------- QKV GEMM: [16384x384] x [384x2304] + bias ----------------
// B tile (64 cols x 384 K, bf16) async-staged in LDS, shared by 8 waves.
__global__ __launch_bounds__(256)
void k_qkv_gemm(const bf16_t* __restrict__ hsb, const bf16_t* __restrict__ wT,
                const float* __restrict__ bias, bf16_t* __restrict__ qout,
                bf16_t* __restrict__ kout, bf16_t* __restrict__ vTout) {
  __shared__ bf16_t ldsB[64 * 384];                // 48 KB
  const int tid = threadIdx.x;
  const int lane = tid & 31, wave = tid >> 5;
  const int m = lane & 15, half = lane >> 4;
  const int row0 = (blockIdx.y * 8 + wave) * 16;   // token tile
  const int col0 = blockIdx.x * 64;                // output-channel tile

  // async stage of B tile: 64*384 bf16 = 3072 x 16B chunks, 12/thread
#pragma unroll
  for (int i = 0; i < 12; ++i) {
    int idx = tid + i * 256;
    int r = idx / 48, off = (idx - r * 48) * 8;
    async_copy16(&ldsB[r * 384 + off], wT + (size_t)(col0 + r) * 384 + off);
  }
  wait_async_0();
  __syncthreads();

  const bf16_t* arow = hsb + (size_t)(row0 + m) * 384;
  v8f acc[4] = {};
#pragma unroll
  for (int kk = 0; kk < 12; ++kk) {
    const int kb = kk * 32;
    v16bf a = a_frag_bf16(arow + kb + half * 16, arow + kb + half * 16 + 8);
#pragma unroll
    for (int t = 0; t < 4; ++t) {
      v16bf b = *(const v16bf*)&ldsB[(t * 16 + m) * 384 + kb + half * 16];
      acc[t] = wmma_bf16(a, b, acc[t]);
    }
  }
#pragma unroll
  for (int t = 0; t < 4; ++t) {
    const int n = col0 + t * 16 + m;
    const float bv = bias[n];
#pragma unroll
    for (int v = 0; v < 8; ++v) {
      const int row = row0 + v + half * 8;
      const float val = acc[t][v] + bv;
      const int b = row >> 12, tok = row & 4095;
      if (n < 768) {                     // Q: token-major
        qout[(size_t)row * 768 + n] = (__bf16)val;
      } else if (n < 1536) {             // K: per-head [key][dim]
        const int c = n - 768, h = c / 96, d = c - h * 96;
        kout[((size_t)(b * 8 + h) * 4096 + tok) * 96 + d] = (__bf16)val;
      } else {                           // V: per-head transposed [dim][key]
        const int c = n - 1536, h = c / 96, d = c - h * 96;
        vTout[((size_t)(b * 8 + h) * 96 + d) * 4096 + tok] = (__bf16)val;
      }
    }
  }
}

// ---------------- Q 2x2 max-pool + scale -> [b*8+h][1024][96] --------------
__global__ __launch_bounds__(256)
void k_pool_q(const bf16_t* __restrict__ qb, bf16_t* __restrict__ qp) {
  int idx = blockIdx.x * 256 + threadIdx.x;
  if (idx >= 4 * 1024 * 768) return;
  int c = idx % 768, r = idx / 768;
  int wq = r & 31, hq = (r >> 5) & 31, b = r >> 10;
  const bf16_t* base = qb + (size_t)b * 4096 * 768;
  int r0 = (2 * hq) * 64 + 2 * wq;
  float v00 = (float)base[(size_t)r0 * 768 + c];
  float v01 = (float)base[(size_t)(r0 + 1) * 768 + c];
  float v10 = (float)base[(size_t)(r0 + 64) * 768 + c];
  float v11 = (float)base[(size_t)(r0 + 65) * 768 + c];
  float mx = fmaxf(fmaxf(v00, v01), fmaxf(v10, v11)) * 0.1020620726159658f; // 96^-0.5
  int h = c / 96, d = c - h * 96;
  qp[((size_t)(b * 8 + h) * 1024 + (hq * 32 + wq)) * 96 + d] = (__bf16)mx;
}

// ---------------- Flash attention: per (b,h), Lq=1024, Lk=4096, d=96 -------
// Double-buffered K/V chunk tiles, async-staged into LDS, shared by 8 waves.
__global__ __launch_bounds__(256)
void k_attention(const bf16_t* __restrict__ qp, const bf16_t* __restrict__ kmat,
                 const bf16_t* __restrict__ vT, bf16_t* __restrict__ aout) {
  __shared__ bf16_t kTile[2][32 * 96];              // 2 x 6 KB  [key][dim]
  __shared__ bf16_t vTile[2][96 * 32];              // 2 x 6 KB  [dim][key]
  __shared__ float pbuf[8][16][33];                 // per-wave P transpose
  const int tid = threadIdx.x;
  const int lane = tid & 31, wave = tid >> 5;
  const int m = lane & 15, half = lane >> 4;
  const int bh = blockIdx.x >> 3;                   // b*8+h
  const int q0 = (blockIdx.x & 7) * 128 + wave * 16;

  const bf16_t* kbase = kmat + (size_t)bh * 4096 * 96;
  const bf16_t* vbase = vT + (size_t)bh * 96 * 4096;

  // issue one stage = 3 async 16B copies per thread; split is wave-aligned
  auto stage = [&](int buf, int key0) {
#pragma unroll
    for (int i = 0; i < 3; ++i) {
      int idx = tid + i * 256;
      if (idx < 384) {
        int r = idx / 12, off = (idx - r * 12) * 8;
        async_copy16(&kTile[buf][r * 96 + off],
                     kbase + (size_t)(key0 + r) * 96 + off);
      } else {
        int cv = idx - 384;
        int r = cv >> 2, off = (cv & 3) * 8;
        async_copy16(&vTile[buf][r * 32 + off],
                     vbase + (size_t)r * 4096 + key0 + off);
      }
    }
  };

  stage(0, 0);                                      // prefetch chunk 0

  const bf16_t* qrow = qp + ((size_t)bh * 1024 + q0 + m) * 96;
  v16bf aq[3];
#pragma unroll
  for (int kk = 0; kk < 3; ++kk)
    aq[kk] = a_frag_bf16(qrow + kk * 32 + half * 16, qrow + kk * 32 + half * 16 + 8);

  v8f O[6] = {};
  float mi[8], li[8];
#pragma unroll
  for (int v = 0; v < 8; ++v) { mi[v] = -3.0e38f; li[v] = 0.f; }

  for (int c = 0; c < 128; ++c) {                   // 32-key chunks
    const int cur = c & 1;
    if (c < 127) {
      stage(cur ^ 1, (c + 1) * 32);                 // overlap next stage
      wait_async_le3();                             // older 3 ops (cur) done
    } else {
      wait_async_0();
    }
    __syncthreads();                                // cur tiles visible

    v8f S0 = {}, S1 = {};
#pragma unroll
    for (int kk = 0; kk < 3; ++kk) {
      v16bf b0 = *(const v16bf*)&kTile[cur][m * 96 + kk * 32 + half * 16];
      v16bf b1 = *(const v16bf*)&kTile[cur][(16 + m) * 96 + kk * 32 + half * 16];
      S0 = wmma_bf16(aq[kk], b0, S0);
      S1 = wmma_bf16(aq[kk], b1, S1);
    }
    // online softmax; C-layout row = v + half*8 spans lanes of a 16-group
    float p0[8], p1[8];
#pragma unroll
    for (int v = 0; v < 8; ++v) {
      float t = fmaxf(S0[v], S1[v]);
#pragma unroll
      for (int off = 1; off < 16; off <<= 1) t = fmaxf(t, __shfl_xor(t, off, 32));
      const float nm = fmaxf(mi[v], t);
      const float alpha = exp2f((mi[v] - nm) * L2E);
      mi[v] = nm;
      p0[v] = exp2f((S0[v] - nm) * L2E);
      p1[v] = exp2f((S1[v] - nm) * L2E);
      float rs = p0[v] + p1[v];
#pragma unroll
      for (int off = 1; off < 16; off <<= 1) rs += __shfl_xor(rs, off, 32);
      li[v] = li[v] * alpha + rs;
#pragma unroll
      for (int t6 = 0; t6 < 6; ++t6) O[t6][v] *= alpha;
    }
    // transpose P (C layout -> A layout) through wave-private LDS
#pragma unroll
    for (int v = 0; v < 8; ++v) {
      pbuf[wave][v + half * 8][m]      = p0[v];
      pbuf[wave][v + half * 8][16 + m] = p1[v];
    }
    __syncthreads();                                // P visible
    v16bf pa;
#pragma unroll
    for (int j = 0; j < 8; ++j) {
      pa[j]     = (__bf16)pbuf[wave][m][half * 16 + j];
      pa[8 + j] = (__bf16)pbuf[wave][m][half * 16 + 8 + j];
    }
#pragma unroll
    for (int t6 = 0; t6 < 6; ++t6) {
      v16bf bv = *(const v16bf*)&vTile[cur][(t6 * 16 + m) * 32 + half * 16];
      O[t6] = wmma_bf16(pa, bv, O[t6]);
    }
    __syncthreads();                                // cur reads done (restage safe)
  }
  const int b = bh >> 3, h = bh & 7;
#pragma unroll
  for (int v = 0; v < 8; ++v) {
    const float inv = 1.f / li[v];
    const int tok = q0 + v + half * 8;
#pragma unroll
    for (int t6 = 0; t6 < 6; ++t6)
      aout[((size_t)b * 1024 + tok) * 768 + h * 96 + t6 * 16 + m] =
          (__bf16)(O[t6][v] * inv);
  }
}

// ---------------- Proj GEMM: [4096x768] x [768x768] + bias -> fp32 out -----
__global__ __launch_bounds__(256)
void k_proj_gemm(const bf16_t* __restrict__ A, const bf16_t* __restrict__ wT,
                 const float* __restrict__ bias, float* __restrict__ out) {
  __shared__ bf16_t ldsB[32 * 768];                // 48 KB
  const int tid = threadIdx.x;
  const int lane = tid & 31, wave = tid >> 5;
  const int m = lane & 15, half = lane >> 4;
  const int row0 = (blockIdx.y * 8 + wave) * 16;
  const int col0 = blockIdx.x * 32;

#pragma unroll
  for (int i = 0; i < 12; ++i) {
    int idx = tid + i * 256;
    int r = idx / 96, off = (idx - r * 96) * 8;
    async_copy16(&ldsB[r * 768 + off], wT + (size_t)(col0 + r) * 768 + off);
  }
  wait_async_0();
  __syncthreads();

  const bf16_t* arow = A + (size_t)(row0 + m) * 768;
  v8f acc[2] = {};
#pragma unroll
  for (int kk = 0; kk < 24; ++kk) {
    const int kb = kk * 32;
    v16bf a = a_frag_bf16(arow + kb + half * 16, arow + kb + half * 16 + 8);
#pragma unroll
    for (int t = 0; t < 2; ++t) {
      v16bf b = *(const v16bf*)&ldsB[(t * 16 + m) * 768 + kb + half * 16];
      acc[t] = wmma_bf16(a, b, acc[t]);
    }
  }
#pragma unroll
  for (int t = 0; t < 2; ++t) {
    const int n = col0 + t * 16 + m;
    const float bv = bias[n];
#pragma unroll
    for (int v = 0; v < 8; ++v)
      out[(size_t)(row0 + v + half * 8) * 768 + n] = acc[t][v] + bv;
  }
}

extern "C" void kernel_launch(void* const* d_in, const int* in_sizes, int n_in,
                              void* d_out, int out_size, void* d_ws, size_t ws_size,
                              hipStream_t stream) {
  const float* hs     = (const float*)d_in[0];
  const float* qkv_w  = (const float*)d_in[1];
  const float* qkv_b  = (const float*)d_in[2];
  const float* proj_w = (const float*)d_in[3];
  const float* proj_b = (const float*)d_in[4];
  float* out = (float*)d_out;
  (void)in_sizes; (void)n_in; (void)out_size; (void)ws_size;

  char* ws = (char*)d_ws;
  size_t off = 0;
  auto take = [&](size_t bytes) -> char* {
    char* p = ws + off;
    off += (bytes + 255) & ~(size_t)255;
    return p;
  };
  bf16_t* hsb    = (bf16_t*)take((size_t)16384 * 384 * 2);  // hs bf16
  bf16_t* wqkvT  = (bf16_t*)take((size_t)2304 * 384 * 2);   // qkv_w^T bf16
  bf16_t* wprojT = (bf16_t*)take((size_t)768 * 768 * 2);    // proj_w^T bf16
  bf16_t* qbuf   = (bf16_t*)take((size_t)16384 * 768 * 2);  // q (pre-pool)
  bf16_t* kbuf   = (bf16_t*)take((size_t)32 * 4096 * 96 * 2);
  bf16_t* vTbuf  = (bf16_t*)take((size_t)32 * 96 * 4096 * 2);
  bf16_t* qpbuf  = (bf16_t*)take((size_t)32 * 1024 * 96 * 2);
  bf16_t* abuf   = (bf16_t*)take((size_t)4096 * 768 * 2);

  k_cvt_bf16<<<(16384 * 384 + 255) / 256, 256, 0, stream>>>(hs, hsb, 16384 * 384);
  k_transpose_bf16<<<(2304 * 384 + 255) / 256, 256, 0, stream>>>(qkv_w, wqkvT, 384, 2304);
  k_transpose_bf16<<<(768 * 768 + 255) / 256, 256, 0, stream>>>(proj_w, wprojT, 768, 768);
  k_qkv_gemm<<<dim3(36, 128), 256, 0, stream>>>(hsb, wqkvT, qkv_b, qbuf, kbuf, vTbuf);
  k_pool_q<<<(4 * 1024 * 768 + 255) / 256, 256, 0, stream>>>(qbuf, qpbuf);
  k_attention<<<256, 256, 0, stream>>>(qpbuf, kbuf, vTbuf, abuf);
  k_proj_gemm<<<dim3(24, 32), 256, 0, stream>>>(abuf, wprojT, proj_b, out);
}
